// GIN_21887153341052
// MI455X (gfx1250) — compile-verified, matrix operand order checked by reading, not run on hardware
//
#include <hip/hip_runtime.h>
#include <hip/hip_bf16.h>

typedef __attribute__((ext_vector_type(16))) __bf16 v16bf;
typedef __attribute__((ext_vector_type(8)))  float  v8f;

#define HDIM 64
#define TDIM 8
#define GDIM 64
#define BN_EPS 1e-5f

// float atomic max via int/uint ordering trick (works with -inf init)
__device__ __forceinline__ void atomicMaxF(float* addr, float val) {
    if (val >= 0.0f) atomicMax((int*)addr, __float_as_int(val));
    else             atomicMin((unsigned int*)addr, __float_as_uint(val));
}

// -------------------------------------------------------------------------
// Y[N,64] = A[N,K] @ W[K,64] + bias   (bf16 WMMA, f32 accumulate)
// Block: 256 threads = 8 waves; each wave computes a 16x64 output slab.
// -------------------------------------------------------------------------
template<int K>
__global__ __launch_bounds__(256) void gemm_bias_wmma(
    const float* __restrict__ A, const float* __restrict__ W,
    const float* __restrict__ bias, float* __restrict__ Y, int N)
{
    // Stage weights in LDS, transposed & converted to bf16: Wt[n*K + k] = W[k,n]
    __shared__ __bf16 Wt[HDIM * K];
    const int tid = threadIdx.x;
    for (int i = tid; i < HDIM * K; i += 256) {
        const int n = i & (HDIM - 1);
        const int k = i >> 6;
        Wt[n * K + k] = (__bf16)W[k * HDIM + n];
    }
    __syncthreads();

    const int wave = tid >> 5;
    const int lane = tid & 31;
    const int g    = lane >> 4;     // half-wave group
    const int lr   = lane & 15;     // row (A) / col (B,D) within 16x16 tile
    const int row0 = blockIdx.x * 128 + wave * 16;

    int arow = row0 + lr;           // clamp so OOB waves still load valid memory
    if (arow > N - 1) arow = N - 1;
    const float* ap = A + (size_t)arow * K + 8 * g;

    v8f acc[4] = {};                // 4 tiles of 16 columns each

#pragma unroll
    for (int kb = 0; kb < K; kb += 32) {
        // A fragment: elems 0..7 <-> K = kb+8g+0..7 ; elems 8..15 <-> K = kb+16+8g+0..7
        v16bf a;
#pragma unroll
        for (int e = 0; e < 8; ++e) {
            a[e]     = (__bf16)ap[kb + e];
            a[e + 8] = (__bf16)ap[kb + 16 + e];
        }
#pragma unroll
        for (int j = 0; j < 4; ++j) {
            const __bf16* bp = &Wt[(size_t)(16 * j + lr) * K + kb + 8 * g];
            v16bf b;
#pragma unroll
            for (int e = 0; e < 8; ++e) { b[e] = bp[e]; b[e + 8] = bp[16 + e]; }
            acc[j] = __builtin_amdgcn_wmma_f32_16x16x32_bf16(
                false, a, false, b, (short)0, acc[j], false, false);
        }
    }

    // D layout: lane = col (lr), VGPR r -> row = 8g + r
#pragma unroll
    for (int j = 0; j < 4; ++j) {
        const int col = 16 * j + lr;
        const float bb = bias[col];
#pragma unroll
        for (int r = 0; r < 8; ++r) {
            const int m = row0 + 8 * g + r;
            if (m < N) Y[(size_t)m * HDIM + col] = acc[j][r] + bb;
        }
    }
}

// -------------------------------------------------------------------------
// Column sums / sum-of-squares for BatchNorm batch statistics.
// stats[0..63] = sum, stats[64..127] = sumsq
// -------------------------------------------------------------------------
__global__ __launch_bounds__(256) void colstats(
    const float* __restrict__ Y, float* __restrict__ stats, int N)
{
    const int c  = threadIdx.x & 63;
    const int rg = threadIdx.x >> 6;     // 0..3 row groups per block
    float s = 0.f, q = 0.f;
    for (int r = blockIdx.x * 4 + rg; r < N; r += gridDim.x * 4) {
        const float v = Y[(size_t)r * HDIM + c];
        s += v; q += v * v;
    }
    __shared__ float ls[256], lq[256];
    ls[threadIdx.x] = s; lq[threadIdx.x] = q;
    __syncthreads();
    if (rg == 0) {
        s = ls[c] + ls[64 + c] + ls[128 + c] + ls[192 + c];
        q = lq[c] + lq[64 + c] + lq[128 + c] + lq[192 + c];
        atomicAdd(&stats[c], s);
        atomicAdd(&stats[HDIM + c], q);
    }
}

// In-place BatchNorm(train, biased var) + ReLU
__global__ __launch_bounds__(256) void bnrelu(
    float* __restrict__ Y, const float* __restrict__ stats,
    const float* __restrict__ gamma, const float* __restrict__ beta, int N)
{
    const size_t i = (size_t)blockIdx.x * 256 + threadIdx.x;
    if (i >= (size_t)N * HDIM) return;
    const int c = (int)(i & (HDIM - 1));
    const float invN = 1.0f / (float)N;
    const float mean = stats[c] * invN;
    const float var  = stats[HDIM + c] * invN - mean * mean;
    const float rstd = rsqrtf(var + BN_EPS);
    const float v = (Y[i] - mean) * rstd * gamma[c] + beta[c];
    Y[i] = v > 0.f ? v : 0.f;
}

// GIN aggregation: agg[dst] += h[src], 64 threads per edge (coalesced row)
__global__ __launch_bounds__(256) void edge_agg(
    const float* __restrict__ h, const int* __restrict__ src,
    const int* __restrict__ dst, float* __restrict__ agg, int E)
{
    const int e = blockIdx.x * 4 + (threadIdx.x >> 6);
    if (e >= E) return;
    const int c = threadIdx.x & 63;
    const int s = src[e], d = dst[e];
    atomicAdd(&agg[(size_t)d * HDIM + c], h[(size_t)s * HDIM + c]);
}

// Per-node readout h@lw + lb, then segment_max over batch via atomicMaxF
__global__ __launch_bounds__(256) void readout(
    const float* __restrict__ h, const float* __restrict__ lw,
    const float* __restrict__ lb, const int* __restrict__ batch,
    float* __restrict__ seg, int N)
{
    const int idx = blockIdx.x * 256 + threadIdx.x;
    if (idx >= N * TDIM) return;
    const int n = idx >> 3, t = idx & 7;
    const float* hp = h + (size_t)n * HDIM;
    float s = lb[t];
#pragma unroll
    for (int k = 0; k < HDIM; ++k) s += hp[k] * lw[k * TDIM + t];
    atomicMaxF(&seg[batch[n] * TDIM + t], s);
}

__global__ void init_out(float* out) {
    if (threadIdx.x < GDIM * TDIM) out[threadIdx.x] = 0.f;
}
__global__ void reset_stats_seg(float* stats, float* seg) {
    const int t = threadIdx.x;
    if (t < 2 * HDIM) stats[t] = 0.f;
    if (t < GDIM * TDIM) seg[t] = -__builtin_inff();
}
__global__ void addseg(float* out, const float* seg) {
    const int t = threadIdx.x;
    if (t < GDIM * TDIM) out[t] += seg[t];
}
__global__ __launch_bounds__(256) void copyk(
    float* __restrict__ dst, const float* __restrict__ src, size_t n)
{
    const size_t i = (size_t)blockIdx.x * 256 + threadIdx.x;
    if (i < n) dst[i] = src[i];
}

// -------------------------------------------------------------------------
extern "C" void kernel_launch(void* const* d_in, const int* in_sizes, int n_in,
                              void* d_out, int out_size, void* d_ws, size_t ws_size,
                              hipStream_t stream) {
    const float* x      = (const float*)d_in[0];
    const int*   ei     = (const int*)d_in[1];     // [2,E]: src row, dst row
    const int*   batch  = (const int*)d_in[2];
    const float* fh_W1  = (const float*)d_in[3];
    const float* fh_b1  = (const float*)d_in[4];
    const float* fh_g1  = (const float*)d_in[5];
    const float* fh_be1 = (const float*)d_in[6];
    const float* fh_W2  = (const float*)d_in[7];
    const float* fh_b2  = (const float*)d_in[8];
    const float* fh_g2  = (const float*)d_in[9];
    const float* fh_be2 = (const float*)d_in[10];
    const float* cv_W1  = (const float*)d_in[11];
    const float* cv_b1  = (const float*)d_in[12];
    const float* cv_g1  = (const float*)d_in[13];
    const float* cv_be1 = (const float*)d_in[14];
    const float* cv_W2  = (const float*)d_in[15];
    const float* cv_b2  = (const float*)d_in[16];
    const float* cv_g2  = (const float*)d_in[17];
    const float* cv_be2 = (const float*)d_in[18];
    const float* lin_W  = (const float*)d_in[19];
    const float* lin_b  = (const float*)d_in[20];

    const int N = in_sizes[2];
    const int E = in_sizes[1] / 2;
    const int NCONV = 3;
    const size_t NH = (size_t)N * HDIM;

    float* B0    = (float*)d_ws;
    float* B1    = B0 + NH;
    float* B2    = B1 + NH;
    float* stats = B2 + NH;        // 128 floats
    float* seg   = stats + 128;    // 512 floats
    float* out   = (float*)d_out;          // [G,T]
    float* hout  = out + GDIM * TDIM;      // final h [N,64]

    const int gemmGrid = (N + 127) / 128;
    const int ewGrid   = (int)((NH + 255) / 256);
    const int roGrid   = (N * TDIM + 255) / 256;
    const int eaGrid   = (E + 3) / 4;

    init_out<<<1, 512, 0, stream>>>(out);

    // ---- input MLP: x[ N,128 ] -> B0 -> B1 ----
    gemm_bias_wmma<128><<<gemmGrid, 256, 0, stream>>>(x, fh_W1, fh_b1, B0, N);
    reset_stats_seg<<<1, 512, 0, stream>>>(stats, seg);
    colstats<<<256, 256, 0, stream>>>(B0, stats, N);
    bnrelu<<<ewGrid, 256, 0, stream>>>(B0, stats, fh_g1, fh_be1, N);
    gemm_bias_wmma<64><<<gemmGrid, 256, 0, stream>>>(B0, fh_W2, fh_b2, B1, N);
    reset_stats_seg<<<1, 512, 0, stream>>>(stats, seg);
    colstats<<<256, 256, 0, stream>>>(B1, stats, N);
    bnrelu<<<ewGrid, 256, 0, stream>>>(B1, stats, fh_g2, fh_be2, N);

    float* h = B1;
    float* spare = B0;

    // layer-0 readout (seg was reset to -inf above)
    readout<<<roGrid, 256, 0, stream>>>(h, lin_W, lin_b, batch, seg, N);
    addseg<<<1, 512, 0, stream>>>(out, seg);

    // ---- GIN conv layers ----
    for (int l = 0; l < NCONV; ++l) {
        // agg = h + segment_sum(h[src] -> dst)   (eps = 0)
        copyk<<<ewGrid, 256, 0, stream>>>(spare, h, NH);
        edge_agg<<<eaGrid, 256, 0, stream>>>(h, ei, ei + E, spare, E);

        gemm_bias_wmma<64><<<gemmGrid, 256, 0, stream>>>(
            spare, cv_W1 + (size_t)l * HDIM * HDIM, cv_b1 + l * HDIM, B2, N);
        reset_stats_seg<<<1, 512, 0, stream>>>(stats, seg);
        colstats<<<256, 256, 0, stream>>>(B2, stats, N);
        bnrelu<<<ewGrid, 256, 0, stream>>>(B2, stats, cv_g1 + l * HDIM, cv_be1 + l * HDIM, N);

        gemm_bias_wmma<64><<<gemmGrid, 256, 0, stream>>>(
            B2, cv_W2 + (size_t)l * HDIM * HDIM, cv_b2 + l * HDIM, spare, N);
        reset_stats_seg<<<1, 512, 0, stream>>>(stats, seg);
        colstats<<<256, 256, 0, stream>>>(spare, stats, N);
        bnrelu<<<ewGrid, 256, 0, stream>>>(spare, stats, cv_g2 + l * HDIM, cv_be2 + l * HDIM, N);

        // swap: new h = spare
        float* t = h; h = spare; spare = t;

        readout<<<roGrid, 256, 0, stream>>>(
            h, lin_W + (size_t)(l + 1) * HDIM * TDIM, lin_b + (l + 1) * TDIM,
            batch, seg, N);
        addseg<<<1, 512, 0, stream>>>(out, seg);
    }

    // final h to output
    copyk<<<ewGrid, 256, 0, stream>>>(hout, h, NH);
}
